// Attention_62019327754556
// MI455X (gfx1250) — matrix operand order repaired and checked by reading, not compiled, not run on hardware
//
#include <hip/hip_runtime.h>

#define B_     16
#define CDIM   256
#define HW_    3136
#define Himg   56
#define INNER_ 512
#define NHEADS 8
#define DHEAD_ 64
#define NPIX   (B_*HW_)   // 50176

typedef __attribute__((ext_vector_type(16))) _Float16 v16h;
typedef __attribute__((ext_vector_type(8)))  _Float16 v8h;
typedef __attribute__((ext_vector_type(8)))  float    v8f;
typedef int v4i __attribute__((vector_size(4 * sizeof(int))));

#if __has_builtin(__builtin_amdgcn_global_load_async_to_lds_b128)
#define HAS_ASYNC_LDS 1
#endif

static __device__ __forceinline__ v16h concat8(v8h lo, v8h hi) {
  return __builtin_shufflevector(lo, hi, 0,1,2,3,4,5,6,7,8,9,10,11,12,13,14,15);
}
// A-fragment: lane<16 -> K {kb..kb+15}; lane>=16 -> K {kb+16..kb+31}
static __device__ __forceinline__ v16h load_afrag(const _Float16* row, int kb, int off8) {
  v8h lo = *(const v8h*)(row + kb + 2*off8);
  v8h hi = *(const v8h*)(row + kb + 2*off8 + 8);
  return concat8(lo, hi);
}
// B-fragment: 16 contiguous K values starting at p
static __device__ __forceinline__ v16h load_bfrag(const _Float16* p) {
  v8h lo = *(const v8h*)(p);
  v8h hi = *(const v8h*)(p + 8);
  return concat8(lo, hi);
}
#define WMMA_F32_F16(a, b, c) \
  __builtin_amdgcn_wmma_f32_16x16x32_f16(false, (a), false, (b), (short)0, (c), false, false)

// 16B global -> LDS copy; async (ASYNCcnt, no VGPR round trip) when available
static __device__ __forceinline__ void copy16_g2l(const _Float16* gsrc, _Float16* ldst) {
#ifdef HAS_ASYNC_LDS
  __builtin_amdgcn_global_load_async_to_lds_b128(
      (__attribute__((address_space(1))) v4i*)gsrc,
      (__attribute__((address_space(3))) v4i*)ldst, 0, 0);
#else
  *(uint4*)ldst = *(const uint4*)gsrc;
#endif
}
static __device__ __forceinline__ void wait_async_all() {
#ifdef HAS_ASYNC_LDS
 #if __has_builtin(__builtin_amdgcn_s_wait_asynccnt)
  __builtin_amdgcn_s_wait_asynccnt(0);
 #else
  asm volatile("s_wait_asynccnt 0x0" ::: "memory");
 #endif
#endif
}
// compiler scheduling fence: keep async issues ahead of the MMA body
static __device__ __forceinline__ void sched_fence() {
  asm volatile("" ::: "memory");
}

// ---------------------------------------------------------------- f32 -> f16
__global__ __launch_bounds__(256) void cvt_kernel(const float* __restrict__ in,
                                                  _Float16* __restrict__ out, int n) {
  int i = blockIdx.x * 256 + threadIdx.x;
  if (i < n) out[i] = (_Float16)in[i];
}

// ------------------------------------------- depthwise 3x3 + BN, NCHW -> [pix][c] f16
__global__ __launch_bounds__(256) void dwbn_kernel(
    const float* __restrict__ x, const float* __restrict__ dw,
    const float* __restrict__ g, const float* __restrict__ bbn,
    const float* __restrict__ mbn, const float* __restrict__ vbn,
    _Float16* __restrict__ Y) {
  __shared__ _Float16 tile[32][33];
  int bid = blockIdx.x;                  // B * (C/32) * (HW/32)
  int pt  = bid % (HW_/32);
  int rem = bid / (HW_/32);
  int ct  = rem % (CDIM/32);
  int b   = rem / (CDIM/32);
  int c0 = ct*32, p0 = pt*32;
  int t  = threadIdx.x;
  int cl = t >> 5;                       // 0..7
  int pl = t & 31;                       // 0..31
  int pix = p0 + pl;
  int py = pix / Himg, px = pix % Himg;
  #pragma unroll
  for (int cc = 0; cc < 4; ++cc) {
    int c = c0 + cl + cc*8;
    const float* xc = x + (size_t)(b*CDIM + c)*HW_;
    float acc = 0.f;
    #pragma unroll
    for (int ky = 0; ky < 3; ++ky) {
      int yy = py + ky - 1;
      if (yy < 0 || yy >= Himg) continue;
      #pragma unroll
      for (int kx = 0; kx < 3; ++kx) {
        int xx = px + kx - 1;
        if (xx < 0 || xx >= Himg) continue;
        acc += xc[yy*Himg + xx] * dw[c*9 + ky*3 + kx];
      }
    }
    float r = (acc - mbn[c]) * rsqrtf(vbn[c] + 1e-5f) * g[c] + bbn[c];
    tile[pl][cl + cc*8] = (_Float16)r;
  }
  __syncthreads();
  for (int idx = t; idx < 32*32; idx += 256) {
    int row = idx >> 5, col = idx & 31;
    Y[(size_t)(b*HW_ + p0 + row)*CDIM + (c0 + col)] = tile[row][col];
  }
}

// ------------------------- P[b,h,pix,d] = W[512x256] x Y[pix][256]
// B tile (64 pixels, shared by all 8 waves) staged in LDS, double-buffered
// in K-chunks of 64 via async global->LDS copies. Pipeline rotated so the
// async copy of chunk c+1 overlaps the WMMAs of chunk c.
__global__ __launch_bounds__(256) void gemm_qkv_kernel(
    const _Float16* __restrict__ W,   // [512][256]
    const _Float16* __restrict__ Y,   // [50176][256]
    _Float16* __restrict__ P) {       // [(b*8+h)*3136+pix][64]
  __shared__ _Float16 Bs[2][64][72];  // 72: 144B row stride, 16B aligned
  int bid = blockIdx.x;               // (512/128) * (50176/64)
  int nt = bid % (NPIX/64);
  int mt = bid / (NPIX/64);
  int t = threadIdx.x;
  int w = t >> 5, lane = t & 31;
  int hlf = lane >> 4, l16 = lane & 15;
  int off8 = hlf * 8;
  int m0 = mt*128 + w*16;
  int n0 = nt*64;
  int srow = t >> 1, sq4 = (t & 1) * 2; // staging: thread -> pixel row, 2x16B pairs

  // stage chunk c into buffer buf: Bs[buf][pix][0..63] = Y[n0+pix][c*64 .. +63]
  auto stage = [&](int c, int buf) {
    const _Float16* src = Y + (size_t)(n0 + srow)*CDIM + c*64 + sq4*8;
    copy16_g2l(src,      &Bs[buf][srow][sq4*8]);
    copy16_g2l(src + 8,  &Bs[buf][srow][sq4*8 + 8]);
  };

  v8f acc[4] = {};
  const _Float16* arow = W + (size_t)(m0 + l16)*CDIM;

  stage(0, 0);
  for (int c = 0; c < 4; ++c) {        // 4 K-chunks of 64
    int buf = c & 1;
    wait_async_all();                  // chunk c resident (staged last iteration)
    __syncthreads();                   // ...in every wave; prior reads of buf^1 done
    if (c + 1 < 4) stage(c + 1, buf ^ 1);
    sched_fence();                     // copy of c+1 flies under the MMA body
    #pragma unroll
    for (int kk = 0; kk < 2; ++kk) {
      v16h a = load_afrag(arow, c*64 + kk*32, off8);
      #pragma unroll
      for (int j = 0; j < 4; ++j) {
        v16h bf = load_bfrag(&Bs[buf][j*16 + l16][kk*32 + hlf*16]);
        acc[j] = WMMA_F32_F16(a, bf, acc[j]);
      }
    }
  }

  int cbase = m0 + hlf*8;              // 8 contiguous channels per lane
  int h = cbase >> 6;
  int dbase = cbase & 63;
  #pragma unroll
  for (int j = 0; j < 4; ++j) {
    int p = n0 + j*16 + l16;
    int b = p / HW_, pix = p % HW_;
    v8h o;
    #pragma unroll
    for (int r = 0; r < 8; ++r) o[r] = (_Float16)acc[j][r];
    *(v8h*)(P + ((size_t)((b*NHEADS + h)*HW_ + pix))*DHEAD_ + dbase) = o;
  }
}

// ----------------------------- windowed attention: one block per (b, head, window)
__global__ __launch_bounds__(128) void attn_kernel(
    const _Float16* __restrict__ Q, const _Float16* __restrict__ K,
    const _Float16* __restrict__ V, const float* __restrict__ pos,
    _Float16* __restrict__ O) {
  __shared__ _Float16 qs[64][72], ksm[64][72], vt[64][72], ps[64][72];
  int bid = blockIdx.x;
  int win = bid & 63;
  int rem = bid >> 6;
  int h = rem & 7, b = rem >> 3;
  int wy = win >> 3, wx = win & 7;
  int t = threadIdx.x;
  const size_t base = (size_t)(b*NHEADS + h) * HW_ * DHEAD_;
  // stage q/k (row-major) and v transposed (vt[d][token]); rows 49..63 zero
  for (int idx = t; idx < 64*8; idx += 128) {
    int i = idx >> 3, ch = idx & 7;
    v8h zvv = {};
    if (i < 49) {
      int pix = (wy*7 + i/7)*Himg + (wx*7 + i%7);
      copy16_g2l(Q + base + (size_t)pix*DHEAD_ + ch*8, &qs[i][ch*8]);
      copy16_g2l(K + base + (size_t)pix*DHEAD_ + ch*8, &ksm[i][ch*8]);
      zvv = *(const v8h*)(V + base + (size_t)pix*DHEAD_ + ch*8);
    } else {
      uint4 z = make_uint4(0,0,0,0);
      *(uint4*)(&qs[i][ch*8])  = z;
      *(uint4*)(&ksm[i][ch*8]) = z;
    }
    #pragma unroll
    for (int e = 0; e < 8; ++e) vt[ch*8 + e][i] = zvv[e];   // transpose scatter
  }
  wait_async_all();
  __syncthreads();
  int w = t >> 5, lane = t & 31;
  int hlf = lane >> 4, l16 = lane & 15;
  int off8 = hlf * 8;
  int m0 = w*16;
  // S = q k^T
  v8f S[4] = {};
  #pragma unroll
  for (int kstep = 0; kstep < 2; ++kstep) {
    int kb = kstep*32;
    v16h a = load_afrag(&qs[m0 + l16][0], kb, off8);
    #pragma unroll
    for (int jt = 0; jt < 4; ++jt) {
      v16h bf = load_bfrag(&ksm[jt*16 + l16][kb + hlf*16]);
      S[jt] = WMMA_F32_F16(a, bf, S[jt]);
    }
  }
  // scale + relative position bias + mask + softmax (row across 16 lanes)
  int irow0 = m0 + hlf*8;
  #pragma unroll
  for (int r = 0; r < 8; ++r) {
    int i = irow0 + r;
    int xi = i/7, yi = i%7;
    float s[4];
    #pragma unroll
    for (int jt = 0; jt < 4; ++jt) {
      int j = jt*16 + l16;
      float val = S[jt][r] * 0.125f;
      if (i < 49 && j < 49) {
        int xj = j/7, yj = j%7;
        int rel = (xj - xi + 6)*13 + (yj - yi + 6);
        val += pos[rel*NHEADS + h];
      } else {
        val = -1e30f;
      }
      s[jt] = val;
    }
    float mx = fmaxf(fmaxf(s[0], s[1]), fmaxf(s[2], s[3]));
    #pragma unroll
    for (int o = 8; o >= 1; o >>= 1) mx = fmaxf(mx, __shfl_xor(mx, o, 32));
    float sum = 0.f;
    #pragma unroll
    for (int jt = 0; jt < 4; ++jt) { s[jt] = __expf(s[jt] - mx); sum += s[jt]; }
    #pragma unroll
    for (int o = 8; o >= 1; o >>= 1) sum += __shfl_xor(sum, o, 32);
    float inv = 1.f / sum;
    #pragma unroll
    for (int jt = 0; jt < 4; ++jt)
      ps[i][jt*16 + l16] = (_Float16)(s[jt] * inv);
  }
  __syncthreads();
  // O = P x V  (B-fragments contiguous thanks to transposed vt)
  v8f Oacc[4] = {};
  #pragma unroll
  for (int kstep = 0; kstep < 2; ++kstep) {
    int kb = kstep*32;
    v16h a = load_afrag(&ps[m0 + l16][0], kb, off8);
    int ktok = kb + hlf*16;
    #pragma unroll
    for (int ntd = 0; ntd < 4; ++ntd) {
      int d = ntd*16 + l16;
      v16h bf = load_bfrag(&vt[d][ktok]);
      Oacc[ntd] = WMMA_F32_F16(a, bf, Oacc[ntd]);
    }
  }
  #pragma unroll
  for (int r = 0; r < 8; ++r) {
    int i = irow0 + r;
    if (i >= 49) continue;
    int pix = (wy*7 + i/7)*Himg + (wx*7 + i%7);
    _Float16* orow = O + base + (size_t)pix*DHEAD_;
    #pragma unroll
    for (int ntd = 0; ntd < 4; ++ntd)
      orow[ntd*16 + l16] = (_Float16)Oacc[ntd][r];
  }
}

// ------------------------- out = Wo[256x512] x AO + bias, write NCHW f32
// K-chunk c (64 channels) == head c, so staging reads contiguous 128B rows.
__global__ __launch_bounds__(256) void outproj_kernel(
    const _Float16* __restrict__ W,   // [256][512]
    const _Float16* __restrict__ A,   // [(b*8+h)*3136+pix][64]
    const float* __restrict__ bias,
    float* __restrict__ out) {        // [b][256][3136]
  __shared__ _Float16 Bs[2][64][72];
  __shared__ float tile[128][65];
  int bid = blockIdx.x;               // (256/128) * (50176/64)
  int nt = bid % (NPIX/64);
  int mt = bid / (NPIX/64);
  int t = threadIdx.x;
  int w = t >> 5, lane = t & 31;
  int hlf = lane >> 4, l16 = lane & 15;
  int off8 = hlf * 8;
  int m0 = mt*128 + w*16;
  int n0 = nt*64;
  int b = n0 / HW_;                   // 64 | 3136, tile stays within one image
  int pix0 = n0 % HW_;
  int srow = t >> 1, sq4 = (t & 1) * 2;

  auto stage = [&](int c, int buf) { // chunk c == head c
    const _Float16* src = A + ((size_t)((b*NHEADS + c)*HW_ + pix0 + srow))*DHEAD_ + sq4*8;
    copy16_g2l(src,     &Bs[buf][srow][sq4*8]);
    copy16_g2l(src + 8, &Bs[buf][srow][sq4*8 + 8]);
  };

  v8f acc[4] = {};
  const _Float16* arow = W + (size_t)(m0 + l16)*INNER_;

  stage(0, 0);
  for (int c = 0; c < 8; ++c) {        // 8 K-chunks of 64 (= heads)
    int buf = c & 1;
    wait_async_all();
    __syncthreads();
    if (c + 1 < 8) stage(c + 1, buf ^ 1);
    sched_fence();
    #pragma unroll
    for (int kk = 0; kk < 2; ++kk) {
      v16h a = load_afrag(arow, c*64 + kk*32, off8);
      #pragma unroll
      for (int j = 0; j < 4; ++j) {
        v16h bf = load_bfrag(&Bs[buf][j*16 + l16][kk*32 + hlf*16]);
        acc[j] = WMMA_F32_F16(a, bf, acc[j]);
      }
    }
  }

  int crow0 = w*16 + hlf*8;           // local channel row in 128-tile
  __syncthreads();                    // all waves done with Bs before tile reuse phase
  #pragma unroll
  for (int j = 0; j < 4; ++j) {
    #pragma unroll
    for (int r = 0; r < 8; ++r)
      tile[crow0 + r][j*16 + l16] = acc[j][r] + bias[m0 + hlf*8 + r];
  }
  __syncthreads();
  for (int idx = t; idx < 128*64; idx += 256) {
    int cl2 = idx >> 6, nl = idx & 63;
    int c = mt*128 + cl2;
    out[(size_t)(b*CDIM + c)*HW_ + pix0 + nl] = tile[cl2][nl];
  }
}

// ---------------------------------------------------------------- launcher
extern "C" void kernel_launch(void* const* d_in, const int* in_sizes, int n_in,
                              void* d_out, int out_size, void* d_ws, size_t ws_size,
                              hipStream_t stream) {
  (void)in_sizes; (void)n_in; (void)out_size; (void)ws_size;
  const float* x = (const float*)d_in[0];
  const float* p_dw[3] = {(const float*)d_in[1], (const float*)d_in[7],  (const float*)d_in[13]};
  const float* p_g [3] = {(const float*)d_in[2], (const float*)d_in[8],  (const float*)d_in[14]};
  const float* p_b [3] = {(const float*)d_in[3], (const float*)d_in[9],  (const float*)d_in[15]};
  const float* p_m [3] = {(const float*)d_in[4], (const float*)d_in[10], (const float*)d_in[16]};
  const float* p_v [3] = {(const float*)d_in[5], (const float*)d_in[11], (const float*)d_in[17]};
  const float* p_pw[3] = {(const float*)d_in[6], (const float*)d_in[12], (const float*)d_in[18]};
  const float* pos_emb = (const float*)d_in[19];
  const float* out_w   = (const float*)d_in[20];
  const float* out_b   = (const float*)d_in[21];
  float* out = (float*)d_out;

  char* ws = (char*)d_ws;
  _Float16* Wf[3];
  Wf[0] = (_Float16*)ws; ws += (size_t)INNER_*CDIM*2;
  Wf[1] = (_Float16*)ws; ws += (size_t)INNER_*CDIM*2;
  Wf[2] = (_Float16*)ws; ws += (size_t)INNER_*CDIM*2;
  _Float16* Wo = (_Float16*)ws; ws += (size_t)CDIM*INNER_*2;
  _Float16* Y  = (_Float16*)ws; ws += (size_t)NPIX*CDIM*2;
  _Float16* QKV[3];
  QKV[0] = (_Float16*)ws; ws += (size_t)NPIX*INNER_*2;
  QKV[1] = (_Float16*)ws; ws += (size_t)NPIX*INNER_*2;
  QKV[2] = (_Float16*)ws; ws += (size_t)NPIX*INNER_*2;
  _Float16* AO = (_Float16*)ws; ws += (size_t)NPIX*INNER_*2;

  for (int p = 0; p < 3; ++p)
    cvt_kernel<<<(INNER_*CDIM + 255)/256, 256, 0, stream>>>(p_pw[p], Wf[p], INNER_*CDIM);
  cvt_kernel<<<(CDIM*INNER_ + 255)/256, 256, 0, stream>>>(out_w, Wo, CDIM*INNER_);

  for (int p = 0; p < 3; ++p) {
    dwbn_kernel<<<B_*(CDIM/32)*(HW_/32), 256, 0, stream>>>(
        x, p_dw[p], p_g[p], p_b[p], p_m[p], p_v[p], Y);
    gemm_qkv_kernel<<<(INNER_/128)*(NPIX/64), 256, 0, stream>>>(Wf[p], Y, QKV[p]);
  }
  attn_kernel<<<B_*NHEADS*64, 128, 0, stream>>>(QKV[0], QKV[1], QKV[2], pos_emb, AO);
  outproj_kernel<<<(CDIM/128)*(NPIX/64), 256, 0, stream>>>(Wo, AO, out_b, out);
}
